// GRIDGRU_51771535786032
// MI455X (gfx1250) — matrix operand order repaired and checked by reading, not compiled, not run on hardware
//
#include <hip/hip_runtime.h>
#include <math.h>

// ---------------------------------------------------------------------------
// GridGRU for MI455X (gfx1250): all GEMMs via v_wmma_f32_16x16x32_bf16,
// double-buffered LDS software pipeline + global_prefetch_b8 L2 warm-up.
// fp32 operands converted (RNE) to bf16 on the fly while staging into LDS.
// ---------------------------------------------------------------------------

typedef __attribute__((ext_vector_type(16))) __bf16 v16bf;
typedef __attribute__((ext_vector_type(8)))  float  v8f;

static __device__ __forceinline__ unsigned short f2bf(float f) {
    unsigned int u = __builtin_bit_cast(unsigned int, f);
    u += 0x7fffu + ((u >> 16) & 1u);   // round-to-nearest-even
    return (unsigned short)(u >> 16);
}

struct B32x8 { uint4 lo; uint4 hi; };

static __device__ __forceinline__ v16bf load_frag(const unsigned short* lds,
                                                  int off_lo, int off_hi) {
    B32x8 r;
    r.lo = *reinterpret_cast<const uint4*>(lds + off_lo);  // 16B aligned
    r.hi = *reinterpret_cast<const uint4*>(lds + off_hi);
    return __builtin_bit_cast(v16bf, r);
}

// Generic WMMA GEMM:  acc = [A | A2](k-split) x B  (+ bias)  -> epilogue MODE
//  MODE 0: C = acc                  (plain store, optional bias)
//  MODE 1: s = sigmoid(acc + gates) ; col<split -> u, else rh = s * ht   (gate)
//  MODE 2: hc = tanh(acc + gates)   ; ht += u*(hc-ht); seq = ht          (scan)
//  MODE 3: hn = x + u*(tanh(acc+gd)-x) -> seq                            (final)
template<int BM, int MODE>
__global__ __launch_bounds__(256)
void gg_wmma_gemm(const float* __restrict__ A, const float* __restrict__ A2,
                  int lda, int lda2, int ksplit,
                  const float* __restrict__ B, int ldb,
                  float* __restrict__ C, int ldc,
                  int K,
                  const float* __restrict__ bias, int bias_limit,
                  int ep_split,
                  const float* __restrict__ e_gates, int e_gstride,
                  float* __restrict__ e_u,
                  float* __restrict__ e_ht,
                  float* __restrict__ e_rh,
                  float* __restrict__ e_seq, int e_seqstride) {
    constexpr int BN  = 128;
    constexpr int KC  = 32;
    constexpr int TPW = BM / 16;          // output 16x16 tiles per wave (8 waves)
    constexpr int ALD = BM * KC / 256;    // A floats staged per thread
    constexpr int BLD = BN * KC / 256;    // B floats staged per thread

    __shared__ __align__(16) unsigned short sA[2][BM * KC];   // [row][k]
    __shared__ __align__(16) unsigned short sB[2][BN * KC];   // [col][k]

    const int tid = threadIdx.x;
    const int w   = tid >> 5;             // wave id (wave32)
    const int l   = tid & 31;             // lane id
    const int bn0 = blockIdx.x * BN;
    const int bm0 = blockIdx.y * BM;

    v8f zero = {0.f, 0.f, 0.f, 0.f, 0.f, 0.f, 0.f, 0.f};
    v8f acc[TPW];
#pragma unroll
    for (int i = 0; i < TPW; ++i) acc[i] = zero;

    float ra[ALD], rb[BLD];

    auto load_regs = [&](int k0) {
#pragma unroll
        for (int j = 0; j < ALD; ++j) {
            int e = tid + j * 256;
            int row = e >> 5, col = e & 31;
            int kg  = k0 + col;
            ra[j] = (kg < ksplit)
                        ? A [(size_t)(bm0 + row) * lda  + kg]
                        : A2[(size_t)(bm0 + row) * lda2 + (kg - ksplit)];
        }
#pragma unroll
        for (int j = 0; j < BLD; ++j) {
            int e = tid + j * 256;
            int kr = e >> 7, cc = e & 127;
            rb[j] = B[(size_t)(k0 + kr) * ldb + (bn0 + cc)];
        }
    };
    auto store_lds = [&](int buf) {
#pragma unroll
        for (int j = 0; j < ALD; ++j) {
            int e = tid + j * 256;
            int row = e >> 5, col = e & 31;
            sA[buf][(row << 5) + col] = f2bf(ra[j]);
        }
#pragma unroll
        for (int j = 0; j < BLD; ++j) {
            int e = tid + j * 256;
            int kr = e >> 7, cc = e & 127;
            sB[buf][(cc << 5) + kr] = f2bf(rb[j]);
        }
    };

    const int nc = K / KC;
    // prologue: stage chunk 0
    load_regs(0);
    store_lds(0);

    for (int c = 0; c < nc; ++c) {
        const int cur = c & 1;
        if (c + 2 < nc) {   // gfx1250 global_prefetch_b8: warm L2 two chunks ahead
            __builtin_prefetch(&B[(size_t)((c + 2) * KC + (tid >> 7)) * ldb +
                                  (bn0 + (tid & 127))], 0, 1);
        }
        if (c + 1 < nc) load_regs((c + 1) * KC);   // prefetch into registers
        __syncthreads();   // publishes LDS[cur]; retires reads of LDS[cur^1]

#pragma unroll
        for (int i = 0; i < TPW; ++i) {
            int t  = w * TPW + i;
            int tm = t >> 3, tn = t & 7;
            int half = l >> 4;                    // lane half selects K octet
            int arow = (tm << 4) + (l & 15);
            v16bf af = load_frag(sA[cur], (arow << 5) + (half << 3),
                                          (arow << 5) + 16 + (half << 3));
            int bcol = (tn << 4) + (l & 15);
            int kb   = half << 3;                 // K octet 8h and 16+8h
            v16bf bf = load_frag(sB[cur], (bcol << 5) + kb, (bcol << 5) + kb + 16);
            acc[i] = __builtin_amdgcn_wmma_f32_16x16x32_bf16(
                         false, af, false, bf, (short)0, acc[i], false, false);
        }
        if (c + 1 < nc) store_lds(cur ^ 1);
    }

    // ---------------- epilogue ----------------
#pragma unroll
    for (int i = 0; i < TPW; ++i) {
        int t  = w * TPW + i;
        int tm = t >> 3, tn = t & 7;
        int gcol    = bn0 + (tn << 4) + (l & 15);
        int rowbase = bm0 + (tm << 4) + ((l >> 4) << 3);
#pragma unroll
        for (int r = 0; r < 8; ++r) {
            int grow = rowbase + r;
            float v  = acc[i][r];
            if (bias && gcol < bias_limit) v += bias[gcol];
            if constexpr (MODE == 0) {
                C[(size_t)grow * ldc + gcol] = v;
            } else if constexpr (MODE == 1) {
                float pre = v;
                if (e_gates) pre += e_gates[(size_t)grow * e_gstride + gcol];
                float s = 1.f / (1.f + __expf(-pre));
                if (gcol < ep_split) {
                    e_u[(size_t)grow * ep_split + gcol] = s;
                } else {
                    size_t ix = (size_t)grow * ep_split + (gcol - ep_split);
                    e_rh[ix] = s * e_ht[ix];
                }
            } else if constexpr (MODE == 2) {
                float pre  = v + e_gates[(size_t)grow * e_gstride + gcol];
                float hc   = tanhf(pre);
                size_t ix  = (size_t)grow * ep_split + gcol;
                float hold = e_ht[ix];
                float hn   = hold + e_u[ix] * (hc - hold);
                e_ht[ix]   = hn;
                e_seq[(size_t)grow * e_seqstride + gcol] = hn;
            } else { // MODE 3
                float pre = v + e_gates[(size_t)grow * e_gstride + gcol];
                size_t ix = (size_t)grow * ep_split + gcol;
                float xv  = e_ht[ix];
                float hn  = xv + e_u[ix] * (tanhf(pre) - xv);
                e_seq[(size_t)grow * e_seqstride + gcol] = hn;
            }
        }
    }
}

// ---------------------------------------------------------------------------
extern "C" void kernel_launch(void* const* d_in, const int* in_sizes, int n_in,
                              void* d_out, int out_size, void* d_ws, size_t ws_size,
                              hipStream_t stream) {
    (void)in_sizes; (void)n_in; (void)out_size; (void)ws_size;
    const float* x      = (const float*)d_in[0];  // (32,512,1024) -> 16384 x 1024
    const float* state  = (const float*)d_in[1];  // 32 x 1024
    const float* weight = (const float*)d_in[2];  // 2048 x 6144
    const float* bias   = (const float*)d_in[3];  // 6144
    float*       out    = (float*)d_out;          // h (16384x1024) ++ final_ht

    constexpr int    T   = 512;
    constexpr size_t NT  = 16384;                 // N*T rows
    constexpr int    LDW = 6144;

    float* ws    = (float*)d_ws;
    float* gates = ws;                            // NT x 3072
    float* ht    = gates + NT * 3072;             // 32 x 1024 (running state)
    float* ubuf  = ht    + 32768;                 // 32 x 1024
    float* rhbuf = ubuf  + 32768;                 // 32 x 1024
    float* htseq = rhbuf + 32768;                 // NT x 1024
    float* ud    = htseq + NT * 1024;             // NT x 1024
    float* rx    = ud    + NT * 1024;             // NT x 1024
    float* gd3   = rx    + NT * 1024;             // NT x 1024

    // ht <- state
    hipMemcpyAsync(ht, state, 32768 * sizeof(float), hipMemcpyDeviceToDevice, stream);

    // G1: gates = x @ W[:D, 0:3072] + bias[:3072]
    gg_wmma_gemm<128, 0><<<dim3(24, 128), 256, 0, stream>>>(
        x, nullptr, 1024, 1024, 1024,
        weight, LDW,
        gates, 3072,
        1024,
        bias, 3072,
        0, nullptr, 0, nullptr, nullptr, nullptr, nullptr, 0);

    // Sequential scan: 2 dependent WMMA GEMMs per step (graph-captured launches)
    for (int t = 0; t < T; ++t) {
        // SA: gg = sigmoid(gates[:,t,:2H] + ht@Whtg); u / rh = r*ht
        gg_wmma_gemm<32, 1><<<dim3(16, 1), 256, 0, stream>>>(
            ht, nullptr, 1024, 1024, 1024,
            weight + (size_t)1024 * LDW, LDW,
            nullptr, 0,
            1024,
            nullptr, 0,
            1024,
            gates + (size_t)t * 3072, 512 * 3072,
            ubuf, ht, rhbuf, nullptr, 0);
        // SB: hc = tanh(gates[:,t,2H:] + rh@Whtc); ht += u*(hc-ht); ht_seq[t]=ht
        gg_wmma_gemm<32, 2><<<dim3(8, 1), 256, 0, stream>>>(
            rhbuf, nullptr, 1024, 1024, 1024,
            weight + (size_t)1024 * LDW + 2048, LDW,
            nullptr, 0,
            1024,
            nullptr, 0,
            1024,
            gates + (size_t)t * 3072 + 2048, 512 * 3072,
            ubuf, ht, nullptr,
            htseq + (size_t)t * 1024, 512 * 1024);
    }

    // G2: [x | ht_seq] @ W[0:2048, 3072:5120] + bias -> ud, rx = rd*x  (MODE 1)
    gg_wmma_gemm<128, 1><<<dim3(16, 128), 256, 0, stream>>>(
        x, htseq, 1024, 1024, 1024,
        weight + 3072, LDW,
        nullptr, 0,
        2048,
        bias + 3072, 2048,
        1024,
        nullptr, 0,
        ud, (float*)x, rx, nullptr, 0);

    // G3: gd3 = ht_seq @ Whd[:, 2D:] + bias[3H+2D:]
    gg_wmma_gemm<128, 0><<<dim3(8, 128), 256, 0, stream>>>(
        htseq, nullptr, 1024, 1024, 1024,
        weight + (size_t)1024 * LDW + 5120, LDW,
        gd3, 1024,
        1024,
        bias + 5120, 1024,
        0, nullptr, 0, nullptr, nullptr, nullptr, nullptr, 0);

    // G4: h = x + ud * (tanh(gd3 + rx @ Wxd[:, 2D:]) - x) -> out
    gg_wmma_gemm<128, 3><<<dim3(8, 128), 256, 0, stream>>>(
        rx, nullptr, 1024, 1024, 1024,
        weight + 5120, LDW,
        nullptr, 0,
        1024,
        nullptr, 0,
        1024,
        gd3, 1024,
        ud, (float*)x, nullptr, out, 1024);

    // final_ht -> tail of output
    hipMemcpyAsync(out + NT * 1024, ht, 32768 * sizeof(float),
                   hipMemcpyDeviceToDevice, stream);
}
